// GCN3_84954453115003
// MI455X (gfx1250) — matrix operand order repaired and checked by reading, not compiled, not run on hardware
//
#include <hip/hip_runtime.h>

// ---------------------------------------------------------------------------
// GCN3: 3-layer GCNConv on MI455X (gfx1250, wave32, WMMA)
//   x:[N,128]  edge_index:[2,E]  edge_weight:[E]
//   layer1/2: 128->128 + relu ; layer3: 128->64
// Dense transforms via v_wmma_f32_16x16x32_bf16 (f32 accumulate), fully
// unrolled K (4 WMMA/tile, 16 WMMA/wave with 4-row-panel B reuse).
// Aggregation via L2-resident global_atomic_add_f32 (feature matrix = 51.2MB
// fits in the 192MB L2, so gather/scatter runs at L2 bandwidth).
// ---------------------------------------------------------------------------

typedef __attribute__((ext_vector_type(16))) __bf16 v16bf;
typedef __attribute__((ext_vector_type(8)))  float  v8f;

#define DIN  128
#define DHID 128
#define DOUTF 64

static __device__ __forceinline__ __bf16 f2bf(float f) { return (__bf16)f; }

// ---------------- utility: zero a float buffer ----------------
__global__ void zero_f32(float* __restrict__ p, long n) {
    long i = (long)blockIdx.x * blockDim.x + threadIdx.x;
    long stride = (long)gridDim.x * blockDim.x;
    for (; i < n; i += stride) p[i] = 0.0f;
}

// ---------------- degree: deg[col[e]] += ew[e] ----------------
__global__ void deg_kernel(const int* __restrict__ col, const float* __restrict__ ew,
                           float* __restrict__ deg, int E) {
    long i = (long)blockIdx.x * blockDim.x + threadIdx.x;
    long stride = (long)gridDim.x * blockDim.x;
    for (; i < E; i += stride) atomicAdd(&deg[col[i]], ew[i]);
}

// ---------------- dinv = (deg+1) > 0 ? rsqrt(deg+1) : 0  (in place) --------
__global__ void dinv_kernel(float* __restrict__ deg, int n) {
    long i = (long)blockIdx.x * blockDim.x + threadIdx.x;
    long stride = (long)gridDim.x * blockDim.x;
    for (; i < n; i += stride) {
        float d = deg[i] + 1.0f;   // self-loop weight 1
        deg[i] = d > 0.0f ? rsqrtf(d) : 0.0f;
    }
}

// ---------------- norm[e] = dinv[row]*ew*dinv[col] ----------------
__global__ void norm_kernel(const int* __restrict__ row, const int* __restrict__ col,
                            const float* __restrict__ ew, const float* __restrict__ dinv,
                            float* __restrict__ norm, int E) {
    long i = (long)blockIdx.x * blockDim.x + threadIdx.x;
    long stride = (long)gridDim.x * blockDim.x;
    for (; i < E; i += stride) norm[i] = dinv[row[i]] * ew[i] * dinv[col[i]];
}

// ---------------- nself[i] = dinv[i]^2 ----------------
__global__ void nself_kernel(const float* __restrict__ dinv, float* __restrict__ nself, int n) {
    long i = (long)blockIdx.x * blockDim.x + threadIdx.x;
    long stride = (long)gridDim.x * blockDim.x;
    for (; i < n; i += stride) nself[i] = dinv[i] * dinv[i];
}

// ---------------- GEMM: D[nrows,DOUT_T] = A[nrows,128] @ W[128,DOUT_T] ------
// blockDim.x = 32*(DOUT_T/16); wave w owns 16-column tile w. Each block
// computes a 64-row panel = 4 tiles of 16 rows, reusing the wave's B
// fragments (all 4 K-blocks held in registers) across the 4 row tiles.
// => 16 v_wmma_f32_16x16x32_bf16 per wave.
//
// Register layouts (ISA 7.12.2, wave32):
//  A (16-bit 16x32): lane l (m=l&15, kb=(l>>4)*8): elem ei -> K = kb+ei+(ei>=8?8:0)
//  B (16-bit 32x16): lane l (n=l&15, kb=(l>>4)*8): elem ei -> K = kb+ei+(ei>=8?8:0)
//  C/D (f32 16x16):  lane l (n=l&15): VGPR j -> M = j + (l>>4)*8
template <int DOUT_T>
__global__ __launch_bounds__(32 * (DOUT_T / 16))
void gemm_wmma_bf16(const float* __restrict__ A, const float* __restrict__ W,
                    float* __restrict__ D, int nrows) {
    const int lane = threadIdx.x & 31;
    const int wave = threadIdx.x >> 5;
    const int col0 = wave * 16;
    const int bn   = lane & 15;
    const int hi   = lane >> 4;

    // --- B fragments for all 4 K-blocks (hoisted; W is tiny & L2-resident) ---
    v16bf bv[4];
#pragma unroll
    for (int kk = 0; kk < 4; ++kk) {
        const float* wp = W + (size_t)(kk * 32 + hi * 8) * DOUT_T + col0 + bn;
#pragma unroll
        for (int ei = 0; ei < 16; ++ei) bv[kk][ei] = f2bf(wp[(ei + (ei >= 8 ? 8 : 0)) * DOUT_T]);
    }

    const int m     = lane & 15;
    const int akb   = hi * 8;
    const int mbase = hi * 8;
    const int row0  = blockIdx.x * 64;

#pragma unroll
    for (int p = 0; p < 4; ++p) {
        const int prow0 = row0 + p * 16;
        const int row   = prow0 + m;
        const int arow  = row < nrows ? row : nrows - 1;   // clamp (never stored)
        const float* ar = A + (size_t)arow * DIN + akb;

        v8f acc = {};
#pragma unroll
        for (int kk = 0; kk < 4; ++kk) {
            const float* arp = ar + kk * 32;
            float4 a0 = *(const float4*)(arp + 0);
            float4 a1 = *(const float4*)(arp + 4);
            float4 a2 = *(const float4*)(arp + 16);
            float4 a3 = *(const float4*)(arp + 20);
            v16bf av;
            av[0]  = f2bf(a0.x); av[1]  = f2bf(a0.y); av[2]  = f2bf(a0.z); av[3]  = f2bf(a0.w);
            av[4]  = f2bf(a1.x); av[5]  = f2bf(a1.y); av[6]  = f2bf(a1.z); av[7]  = f2bf(a1.w);
            av[8]  = f2bf(a2.x); av[9]  = f2bf(a2.y); av[10] = f2bf(a2.z); av[11] = f2bf(a2.w);
            av[12] = f2bf(a3.x); av[13] = f2bf(a3.y); av[14] = f2bf(a3.z); av[15] = f2bf(a3.w);

            acc = __builtin_amdgcn_wmma_f32_16x16x32_bf16(
                /*neg_a=*/false, av, /*neg_b=*/false, bv[kk],
                /*c_mod=*/(short)0, acc, /*reuse_a=*/false, /*reuse_b=*/false);
        }

        // --- store 16x16 tile; branch-free fast path for full tiles ---
        float* dp = D + (size_t)prow0 * DOUT_T + col0 + bn;
        if (prow0 + 16 <= nrows) {
#pragma unroll
            for (int j = 0; j < 8; ++j)
                dp[(size_t)(mbase + j) * DOUT_T] = acc[j];
        } else {
#pragma unroll
            for (int j = 0; j < 8; ++j) {
                if (prow0 + mbase + j < nrows)
                    dp[(size_t)(mbase + j) * DOUT_T] = acc[j];
            }
        }
    }
}

// ---------------- edge scatter: Agg[col] += T[row] * norm ----------------
// one wave per edge; lane owns DOUT_T/32 consecutive features (constexpr)
template <int DOUT_T>
__global__ void agg_kernel(const int* __restrict__ row, const int* __restrict__ col,
                           const float* __restrict__ norm, const float* __restrict__ T,
                           float* __restrict__ Agg, int E) {
    constexpr int FPL = DOUT_T / 32;   // 4 or 2
    long g = (long)blockIdx.x * blockDim.x + threadIdx.x;
    long stride = (long)gridDim.x * blockDim.x;
    long total = (long)E * 32;
    for (; g < total; g += stride) {
        int e    = (int)(g >> 5);
        int lane = (int)(g & 31);
        int r = row[e];
        int c = col[e];
        float w = norm[e];
        int f0 = lane * FPL;
        const float* ts = T + (size_t)r * DOUT_T + f0;   // contiguous -> b128/b64
        float* as = Agg + (size_t)c * DOUT_T + f0;
#pragma unroll
        for (int f = 0; f < FPL; ++f) atomicAdd(&as[f], ts[f] * w);
    }
}

// ---------------- epilogue: out = [relu](agg + nself*tmp + bias), float4 ----
template <int LDOUT>
__global__ void epilogue_kernel(const float4* __restrict__ Agg, const float4* __restrict__ T,
                                const float* __restrict__ nself, const float4* __restrict__ bias,
                                float4* __restrict__ Out, int n, int relu) {
    constexpr int DV = (1 << LDOUT) / 4;   // float4s per row
    long g = (long)blockIdx.x * blockDim.x + threadIdx.x;
    long stride = (long)gridDim.x * blockDim.x;
    long total = (long)n * DV;
    for (; g < total; g += stride) {
        int node = (int)(g >> (LDOUT - 2));
        int fv   = (int)(g & (DV - 1));
        float4 a = Agg[g];
        float4 t = T[g];
        float4 b = bias[fv];
        float  s = nself[node];
        float4 v;
        v.x = a.x + s * t.x + b.x;
        v.y = a.y + s * t.y + b.y;
        v.z = a.z + s * t.z + b.z;
        v.w = a.w + s * t.w + b.w;
        if (relu) {
            v.x = fmaxf(v.x, 0.0f); v.y = fmaxf(v.y, 0.0f);
            v.z = fmaxf(v.z, 0.0f); v.w = fmaxf(v.w, 0.0f);
        }
        Out[g] = v;
    }
}

// ---------------------------------------------------------------------------
extern "C" void kernel_launch(void* const* d_in, const int* in_sizes, int n_in,
                              void* d_out, int out_size, void* d_ws, size_t ws_size,
                              hipStream_t stream) {
    const float* x  = (const float*)d_in[0];
    const int*   ei = (const int*)  d_in[1];
    const float* ew = (const float*)d_in[2];
    const float* W1 = (const float*)d_in[3];
    const float* b1 = (const float*)d_in[4];
    const float* W2 = (const float*)d_in[5];
    const float* b2 = (const float*)d_in[6];
    const float* W3 = (const float*)d_in[7];
    const float* b3 = (const float*)d_in[8];
    float* out = (float*)d_out;

    const int N = in_sizes[0] / DIN;     // 100000
    const int E = in_sizes[2];           // 1600000
    const int* erow = ei;                // edge_index[0]
    const int* ecol = ei + E;            // edge_index[1]

    // ---- workspace carving (256B aligned) ----
    char* ws = (char*)d_ws;
    size_t off = 0;
    auto carve = [&](size_t bytes) -> float* {
        float* p = (float*)(ws + off);
        off += (bytes + 255) & ~(size_t)255;
        return p;
    };
    float* dinv  = carve((size_t)N * 4);             // deg, then dinv in place
    float* nself = carve((size_t)N * 4);
    float* norm  = carve((size_t)E * 4);
    float* tmp   = carve((size_t)N * DHID * 4);      // GEMM output
    float* agg   = carve((size_t)N * DHID * 4);      // scatter accumulator
    float* hbuf  = carve((size_t)N * DHID * 4);      // activations between layers
    (void)ws_size; (void)n_in; (void)out_size;

    const int TB = 256;
    auto blk = [](long n, int tb) { return (unsigned)((n + tb - 1) / tb); };

    // ---- normalization precompute ----
    zero_f32<<<blk(N, TB), TB, 0, stream>>>(dinv, N);
    deg_kernel<<<blk(E, TB), TB, 0, stream>>>(ecol, ew, dinv, E);
    dinv_kernel<<<blk(N, TB), TB, 0, stream>>>(dinv, N);
    norm_kernel<<<blk(E, TB), TB, 0, stream>>>(erow, ecol, ew, dinv, norm, E);
    nself_kernel<<<blk(N, TB), TB, 0, stream>>>(dinv, nself, N);

    const unsigned gemm_grid = blk(N, 64);           // 64-row panels
    const long agg_threads = (long)E * 32;

    // ---- layer 1: x(128) -> hbuf(128), relu ----
    gemm_wmma_bf16<DHID><<<gemm_grid, 32 * (DHID / 16), 0, stream>>>(x, W1, tmp, N);
    zero_f32<<<blk((long)N * DHID, TB), TB, 0, stream>>>(agg, (long)N * DHID);
    agg_kernel<DHID><<<blk(agg_threads, TB), TB, 0, stream>>>(erow, ecol, norm, tmp, agg, E);
    epilogue_kernel<7><<<blk((long)N * DHID / 4, TB), TB, 0, stream>>>(
        (const float4*)agg, (const float4*)tmp, nself, (const float4*)b1, (float4*)hbuf, N, 1);

    // ---- layer 2: hbuf(128) -> hbuf(128), relu ----
    gemm_wmma_bf16<DHID><<<gemm_grid, 32 * (DHID / 16), 0, stream>>>(hbuf, W2, tmp, N);
    zero_f32<<<blk((long)N * DHID, TB), TB, 0, stream>>>(agg, (long)N * DHID);
    agg_kernel<DHID><<<blk(agg_threads, TB), TB, 0, stream>>>(erow, ecol, norm, tmp, agg, E);
    epilogue_kernel<7><<<blk((long)N * DHID / 4, TB), TB, 0, stream>>>(
        (const float4*)agg, (const float4*)tmp, nself, (const float4*)b2, (float4*)hbuf, N, 1);

    // ---- layer 3: hbuf(128) -> out(64), no relu ----
    gemm_wmma_bf16<DOUTF><<<gemm_grid, 32 * (DOUTF / 16), 0, stream>>>(hbuf, W3, tmp, N);
    zero_f32<<<blk((long)N * DOUTF, TB), TB, 0, stream>>>(agg, (long)N * DOUTF);
    agg_kernel<DOUTF><<<blk(agg_threads, TB), TB, 0, stream>>>(erow, ecol, norm, tmp, agg, E);
    epilogue_kernel<6><<<blk((long)N * DOUTF / 4, TB), TB, 0, stream>>>(
        (const float4*)agg, (const float4*)tmp, nself, (const float4*)b3, (float4*)out, N, 0);
}
